// ResidualMemcodes_43645457662712
// MI455X (gfx1250) — compile-verified, matrix operand order checked by reading, not run on hardware
//
#include <hip/hip_runtime.h>

// Problem constants (match reference)
#define QQ   4
#define BB   8
#define NN   4096
#define DIMM 1024
#define HH   8
#define CC   1024
#define DHH  128
#define RSCALE 0.08838834764831845f  // 128^-0.5

typedef __attribute__((ext_vector_type(16))) _Float16 v16h;
typedef __attribute__((ext_vector_type(8)))  _Float16 v8h;
typedef __attribute__((ext_vector_type(8)))  float    v8f;

// ---------------------------------------------------------------------------
// Load a 16x32 f16 WMMA A/B fragment from a row-major [16][rowstride] tile.
// ISA layout (CDNA5 ISA, 16-bit A 16x32):
//   lane<16 : row=lane,   halves 0..7 -> K=kbase+0..7,   8..15 -> K=kbase+16..23
//   lane>=16: row=lane-16, halves 0..7 -> K=kbase+8..15, 8..15 -> K=kbase+24..31
// B (32x16) mirrors this: loading the 16 code rows (B^T row-major) with the
// same pattern yields the B operand.
// ---------------------------------------------------------------------------
__device__ __forceinline__ v16h load_frag_f16(const _Float16* __restrict__ base,
                                              int rowstride, int lane, int kbase) {
  const int r    = lane & 15;
  const int koff = (lane >> 4) << 3;  // 0 or 8
  const _Float16* p = base + (size_t)r * rowstride + kbase + koff;
  v8h lo = *(const v8h*)(p);       // K = kbase+koff .. +7
  v8h hi = *(const v8h*)(p + 16);  // K = kbase+16+koff .. +7
  v16h o;
#pragma unroll
  for (int i = 0; i < 8; ++i) { o[i] = lo[i]; o[i + 8] = hi[i]; }
  return o;
}

// Build four K=32 A-fragments (q = (x - out)*SCALE, f32 -> f16) for one token tile.
template <int LAYER>
__device__ __forceinline__ void build_a(v16h (&a)[4], const float* __restrict__ xrow,
                                        const float* __restrict__ orow, int koff) {
#pragma unroll
  for (int ks = 0; ks < 4; ++ks) {
    const float* p  = xrow + ks * 32 + koff;
    const float* po = orow + ks * 32 + koff;
#pragma unroll
    for (int j = 0; j < 8; ++j) {
      float lo = p[j];
      float hi = p[16 + j];
      if (LAYER > 0) { lo -= po[j]; hi -= po[16 + j]; }
      a[ks][j]     = (_Float16)(lo * RSCALE);
      a[ks][j + 8] = (_Float16)(hi * RSCALE);
    }
  }
}

__device__ __forceinline__ void argmax_update(float (&mv)[8], int (&mi)[8],
                                              const v8f& acc, int cidx) {
#pragma unroll
  for (int r = 0; r < 8; ++r) {
    const float v = acc[r];  // C layout: vgpr r -> row r (lanes 0-15) / r+8 (16-31)
    if (v > mv[r]) { mv[r] = v; mi[r] = cidx; }
  }
}

// Half-wave argmax reduction + V gather + output accumulate for one token tile.
template <int LAYER>
__device__ __forceinline__ void reduce_and_write(float (&mv)[8], int (&mi)[8],
                                                 int lane, int tbase, int h,
                                                 const float* __restrict__ Vh,
                                                 float* __restrict__ outq,
                                                 int* __restrict__ outidx) {
#pragma unroll
  for (int m = 8; m >= 1; m >>= 1) {  // xor masks 8,4,2,1 stay within each half-wave
#pragma unroll
    for (int r = 0; r < 8; ++r) {
      const float ov = __shfl_xor(mv[r], m, 32);
      const int   oi = __shfl_xor(mi[r], m, 32);
      if (ov > mv[r] || (ov == mv[r] && oi < mi[r])) { mv[r] = ov; mi[r] = oi; }
    }
  }
  const int half = lane >> 4;
  const int col  = lane & 15;
#pragma unroll
  for (int r = 0; r < 8; ++r) {
    const int t   = tbase + r + half * 8;
    const int idx = mi[r];
    if (col == 0) {
      const int b = t / NN, n = t % NN;
      outidx[(((size_t)b * HH + h) * NN + n) * QQ + LAYER] = idx;
    }
    const float* vsrc = Vh + (size_t)idx * DHH + col * 8;
    float*       dst  = outq + (size_t)t * DIMM + h * DHH + col * 8;
#pragma unroll
    for (int j = 0; j < 8; ++j) {
      float val = vsrc[j];
      if (LAYER > 0) val += dst[j];
      dst[j] = val;
    }
  }
}

// ---------------------------------------------------------------------------
// K/V precompute:  K_f16[q,h,c,:] = codes[q,h,c,:] @ wk[q,h]   (stored f16)
//                  V_f32[q,h,c,:] = codes[q,h,c,:] @ wv[q,h]   (stored f32)
// ---------------------------------------------------------------------------
__global__ __launch_bounds__(DHH) void kv_precompute_kernel(
    const float* __restrict__ codes, const float* __restrict__ wk,
    const float* __restrict__ wv, _Float16* __restrict__ Km, float* __restrict__ Vm) {
  const int qh = blockIdx.x >> 10;        // [0, QQ*HH)
  const int c  = blockIdx.x & (CC - 1);
  const int d  = threadIdx.x;
  const float* crow = codes + ((size_t)qh * CC + c) * DHH;
  const float* wkm  = wk + (size_t)qh * DHH * DHH;
  const float* wvm  = wv + (size_t)qh * DHH * DHH;
  float ak = 0.f, av = 0.f;
#pragma unroll 4
  for (int e = 0; e < DHH; ++e) {
    const float ce = crow[e];
    ak = fmaf(ce, wkm[(size_t)e * DHH + d], ak);
    av = fmaf(ce, wvm[(size_t)e * DHH + d], av);
  }
  const size_t o = ((size_t)qh * CC + c) * DHH + d;
  Km[o] = (_Float16)ak;
  Vm[o] = av;
}

// ---------------------------------------------------------------------------
// One Memcodes layer. Grid: (B*N/128, H). Block: 128 threads = 4 waves.
// Each wave: 32 tokens (two 16-row A tiles sharing every B fragment) of one
// head. B fragments are double-buffered in registers with a 2x-unrolled
// code-tile loop so global_load_b128s for tile ct+1 overlap the 8 WMMAs on
// tile ct (partial s_wait_loadcnt instead of wait-to-zero).
// ---------------------------------------------------------------------------
template <int LAYER>
__global__ __launch_bounds__(128) void memcodes_layer_kernel(
    const float* __restrict__ x, const _Float16* __restrict__ Km,
    const float* __restrict__ Vm, float* __restrict__ outq,
    int* __restrict__ outidx) {
  const int lane = threadIdx.x & 31;
  const int wave = threadIdx.x >> 5;
  const int h    = blockIdx.y;
  const int t0   = blockIdx.x * 128 + wave * 32;  // flat token base over B*N
  const int row  = lane & 15;
  const int koff = (lane >> 4) << 3;

  // A fragments for the two token tiles (rows t0..t0+15 and t0+16..t0+31).
  v16h a0[4], a1[4];
  {
    const int tk0 = t0 + row;
    const int tk1 = t0 + 16 + row;
    build_a<LAYER>(a0, x + (size_t)tk0 * DIMM + h * DHH,
                   outq + (size_t)tk0 * DIMM + h * DHH, koff);
    build_a<LAYER>(a1, x + (size_t)tk1 * DIMM + h * DHH,
                   outq + (size_t)tk1 * DIMM + h * DHH, koff);
  }

  const _Float16* Kh = Km + (size_t)(LAYER * HH + h) * CC * DHH;

  float mv0[8], mv1[8];
  int   mi0[8], mi1[8];
#pragma unroll
  for (int r = 0; r < 8; ++r) {
    mv0[r] = -3.0e38f; mv1[r] = -3.0e38f; mi0[r] = 0; mi1[r] = 0;
  }

  // Ping-pong B buffers; prologue fills bc with tile 0.
  v16h bc[4], bn[4];
#pragma unroll
  for (int ks = 0; ks < 4; ++ks) bc[ks] = load_frag_f16(Kh, DHH, lane, ks * 32);

  for (int ct = 0; ct < CC / 16; ct += 2) {
    // Prefetch tile ct+1 into bn while computing on bc (tile ct).
    {
      const _Float16* kt = Kh + (size_t)(ct + 1) * 16 * DHH;
#pragma unroll
      for (int ks = 0; ks < 4; ++ks) bn[ks] = load_frag_f16(kt, DHH, lane, ks * 32);
    }
    {
      v8f acc0 = {}, acc1 = {};
#pragma unroll
      for (int ks = 0; ks < 4; ++ks) {
        acc0 = __builtin_amdgcn_wmma_f32_16x16x32_f16(false, a0[ks], false, bc[ks],
                                                      (short)0, acc0, false, false);
        acc1 = __builtin_amdgcn_wmma_f32_16x16x32_f16(false, a1[ks], false, bc[ks],
                                                      (short)0, acc1, false, false);
      }
      const int cidx = ct * 16 + (lane & 15);
      argmax_update(mv0, mi0, acc0, cidx);
      argmax_update(mv1, mi1, acc1, cidx);
    }
    // Prefetch tile ct+2 into bc (wraps harmlessly on the last iteration)
    // while computing on bn (tile ct+1).
    {
      const _Float16* kt = Kh + (size_t)((ct + 2) & (CC / 16 - 1)) * 16 * DHH;
#pragma unroll
      for (int ks = 0; ks < 4; ++ks) bc[ks] = load_frag_f16(kt, DHH, lane, ks * 32);
    }
    {
      v8f acc0 = {}, acc1 = {};
#pragma unroll
      for (int ks = 0; ks < 4; ++ks) {
        acc0 = __builtin_amdgcn_wmma_f32_16x16x32_f16(false, a0[ks], false, bn[ks],
                                                      (short)0, acc0, false, false);
        acc1 = __builtin_amdgcn_wmma_f32_16x16x32_f16(false, a1[ks], false, bn[ks],
                                                      (short)0, acc1, false, false);
      }
      const int cidx = (ct + 1) * 16 + (lane & 15);
      argmax_update(mv0, mi0, acc0, cidx);
      argmax_update(mv1, mi1, acc1, cidx);
    }
  }

  const float* Vh = Vm + (size_t)(LAYER * HH + h) * CC * DHH;
  reduce_and_write<LAYER>(mv0, mi0, lane, t0, h, Vh, outq, outidx);
  reduce_and_write<LAYER>(mv1, mi1, lane, t0 + 16, h, Vh, outq, outidx);
}

// ---------------------------------------------------------------------------
extern "C" void kernel_launch(void* const* d_in, const int* in_sizes, int n_in,
                              void* d_out, int out_size, void* d_ws, size_t ws_size,
                              hipStream_t stream) {
  (void)in_sizes; (void)n_in; (void)out_size; (void)ws_size;
  const float* x     = (const float*)d_in[0];
  const float* codes = (const float*)d_in[1];
  const float* wk    = (const float*)d_in[2];
  const float* wv    = (const float*)d_in[3];

  float* outq   = (float*)d_out;                                      // [B,N,DIM] f32
  int*   outidx = (int*)((float*)d_out + (size_t)BB * NN * DIMM);     // [B,H,N,Q]

  const size_t kv_elems = (size_t)QQ * HH * CC * DHH;                 // 4M
  _Float16* Km = (_Float16*)d_ws;                                     // 8 MB f16
  float*    Vm = (float*)((char*)d_ws + kv_elems * sizeof(_Float16)); // 16 MB f32

  // 1) K/V precompute (K -> f16 for WMMA B operands, V -> f32 for gather)
  kv_precompute_kernel<<<QQ * HH * CC, DHH, 0, stream>>>(codes, wk, wv, Km, Vm);

  // 2) Four residual-VQ layers; stream order provides the residual dependency.
  dim3 grid(BB * NN / 128, HH);
  memcodes_layer_kernel<0><<<grid, 128, 0, stream>>>(x, Km, Vm, outq, outidx);
  memcodes_layer_kernel<1><<<grid, 128, 0, stream>>>(x, Km, Vm, outq, outidx);
  memcodes_layer_kernel<2><<<grid, 128, 0, stream>>>(x, Km, Vm, outq, outidx);
  memcodes_layer_kernel<3><<<grid, 128, 0, stream>>>(x, Km, Vm, outq, outidx);
}
